// NeuroSymbolicAttention_16681652977756
// MI455X (gfx1250) — matrix operand order repaired and checked by reading, not compiled, not run on hardware
//
#include <hip/hip_runtime.h>
#include <hip/hip_bf16.h>
#include <math.h>

// Problem constants
#define B_  2
#define S_  2048
#define D_  512
#define H_  8
#define HD_ 64
#define NEGV  (-1000000000.0f)
#define SCALE (0.125f)   // 1/sqrt(64)

typedef __attribute__((ext_vector_type(16))) _Float16 v16h;
typedef __attribute__((ext_vector_type(8)))  float    v8f;
typedef __attribute__((ext_vector_type(4)))  float    f32x4;

// ---------------------------------------------------------------------------
// WMMA fragment helpers (contiguous k order k = 16*(L/16) + e, wave32)
// ---------------------------------------------------------------------------

// A fragment: 16x32 f16, row-major source with leading dim lda.
// lane L holds row m0 + L%16; element e -> k = k0 + (L/16)*16 + e
static __device__ inline v16h frag_a_f16(const _Float16* __restrict__ A, int lda,
                                         int m0, int k0, int lane) {
  v16h a;
  const _Float16* row = A + (size_t)(m0 + (lane & 15)) * lda;
  int kb = k0 + ((lane >> 4) << 4);
#pragma unroll
  for (int e = 0; e < 8; ++e) a[e]     = row[kb + e];
#pragma unroll
  for (int e = 0; e < 8; ++e) a[8 + e] = row[kb + 8 + e];
  return a;
}

// A fragment sourced from f32 LDS (attention probabilities), converted to f16.
static __device__ inline v16h frag_a_f32(const float* __restrict__ A, int lda,
                                         int m0, int k0, int lane) {
  v16h a;
  const float* row = A + (size_t)(m0 + (lane & 15)) * lda;
  int kb = k0 + ((lane >> 4) << 4);
#pragma unroll
  for (int e = 0; e < 8; ++e) a[e]     = (_Float16)row[kb + e];
#pragma unroll
  for (int e = 0; e < 8; ++e) a[8 + e] = (_Float16)row[kb + 8 + e];
  return a;
}

// B fragment: 32x16 f16 from "n-major" storage Bt[n][k] (contiguous in k).
// lane L holds column n0 + L%16; element e -> k = k0 + (L/16)*16 + e
static __device__ inline v16h frag_b_nmajor(const _Float16* __restrict__ Bt, int ldb,
                                            int k0, int n0, int lane) {
  v16h b;
  const _Float16* row = Bt + (size_t)(n0 + (lane & 15)) * ldb;
  int kb = k0 + ((lane >> 4) << 4);
#pragma unroll
  for (int e = 0; e < 16; ++e) b[e] = row[kb + e];
  return b;
}

// B fragment from k-major storage Bm[k][n] (column strided).
static __device__ inline v16h frag_b_kmajor(const _Float16* __restrict__ Bm, int ldb,
                                            int k0, int n0, int lane) {
  v16h b;
  int n  = n0 + (lane & 15);
  int kb = k0 + ((lane >> 4) << 4);
#pragma unroll
  for (int e = 0; e < 16; ++e) b[e] = Bm[(size_t)(kb + e) * ldb + n];
  return b;
}

static __device__ inline v8f wmma32(v16h a, v16h b, v8f c) {
  return __builtin_amdgcn_wmma_f32_16x16x32_f16(false, a, false, b, (short)0, c,
                                                false, false);
}

// ---------------------------------------------------------------------------
// Generic GEMM: C(MxN) = A(MxK) * W(KxN) + bias, A from f32 (optionally split
// concat of two sources along K), W f32 row-major.
// MODE 0: f16 out + bias; MODE 1: f32 out + bias; MODE 2: f32 sigmoid(x+bias).
// Block = 128 threads (4 waves), 64x64 C tile, each wave does a 32x32 subtile.
// ---------------------------------------------------------------------------
template <int MODE>
__global__ __launch_bounds__(128) void gemm64(
    const float* __restrict__ A0, const float* __restrict__ A1, int Ksplit, int lda,
    const float* __restrict__ W, int ldw, const float* __restrict__ bias,
    _Float16* __restrict__ outh, float* __restrict__ outf,
    int N, int K) {
  __shared__ _Float16 As[64][40];
  __shared__ _Float16 Bs[64][40];

  const int tid  = threadIdx.x;
  const int lane = tid & 31;
  const int wave = tid >> 5;
  const int wm   = (wave >> 1) * 32;
  const int wn   = (wave & 1) * 32;
  const int m0   = blockIdx.y * 64;
  const int n0   = blockIdx.x * 64;

  v8f acc[2][2] = {};

  for (int k0 = 0; k0 < K; k0 += 32) {
    // Stage A tile 64x32 (f32 -> f16)
#pragma unroll
    for (int i = 0; i < 16; ++i) {
      int idx = tid + i * 128;
      int r = idx >> 5, c = idx & 31;
      int k = k0 + c;
      float v = (k < Ksplit) ? A0[(size_t)(m0 + r) * lda + k]
                             : A1[(size_t)(m0 + r) * lda + (k - Ksplit)];
      As[r][c] = (_Float16)v;
    }
    // Stage W tile 32x64 transposed into n-major Bs[n][k]
#pragma unroll
    for (int i = 0; i < 16; ++i) {
      int idx = tid + i * 128;
      int n = idx & 63, k = idx >> 6;
      Bs[n][k] = (_Float16)W[(size_t)(k0 + k) * ldw + (n0 + n)];
    }
    __syncthreads();

    v16h aT[2], bT[2];
    aT[0] = frag_a_f16(&As[0][0], 40, wm, 0, lane);
    aT[1] = frag_a_f16(&As[0][0], 40, wm + 16, 0, lane);
    bT[0] = frag_b_nmajor(&Bs[0][0], 40, 0, wn, lane);
    bT[1] = frag_b_nmajor(&Bs[0][0], 40, 0, wn + 16, lane);
#pragma unroll
    for (int ti = 0; ti < 2; ++ti)
#pragma unroll
      for (int tj = 0; tj < 2; ++tj)
        acc[ti][tj] = wmma32(aT[ti], bT[tj], acc[ti][tj]);
    __syncthreads();
  }

  // Epilogue (branch-free per MODE via template instantiation)
#pragma unroll
  for (int ti = 0; ti < 2; ++ti)
#pragma unroll
    for (int tj = 0; tj < 2; ++tj) {
      int n = n0 + wn + tj * 16 + (lane & 15);
      float bv = bias[n];
#pragma unroll
      for (int r = 0; r < 8; ++r) {
        int m = m0 + wm + ti * 16 + r + ((lane >> 4) << 3);
        float v = acc[ti][tj][r] + bv;
        size_t idx = (size_t)m * N + n;
        if (MODE == 0)       outh[idx] = (_Float16)v;
        else if (MODE == 1)  outf[idx] = v;
        else                 outf[idx] = 1.0f / (1.0f + __expf(-v));
      }
    }
}

// ---------------------------------------------------------------------------
// gate[m] = mean over D of gbuf[m][:]  (gbuf already holds sigmoid values)
// ---------------------------------------------------------------------------
__global__ __launch_bounds__(128) void gate_reduce(const float* __restrict__ gbuf,
                                                   float* __restrict__ gate) {
  __shared__ float red[128];
  int m = blockIdx.x;
  float s = 0.0f;
  for (int i = threadIdx.x; i < D_; i += 128) s += gbuf[(size_t)m * D_ + i];
  red[threadIdx.x] = s;
  __syncthreads();
  for (int st = 64; st > 0; st >>= 1) {
    if (threadIdx.x < st) red[threadIdx.x] += red[threadIdx.x + st];
    __syncthreads();
  }
  if (threadIdx.x == 0) gate[m] = red[0] * (1.0f / (float)D_);
}

// ---------------------------------------------------------------------------
// rel[b][h][s] = sum_d relation_weights[b][s][d] * Wr[d][h] + br[h]
// ---------------------------------------------------------------------------
__global__ __launch_bounds__(64) void rel_kernel(const float* __restrict__ rw,
                                                 const float* __restrict__ Wr,
                                                 const float* __restrict__ br,
                                                 float* __restrict__ rel) {
  __shared__ float row[D_];
  int bs = blockIdx.x;
  int b = bs / S_, s = bs % S_;
  const float* src = rw + (size_t)bs * D_;
  for (int i = threadIdx.x; i < D_; i += 64) row[i] = src[i];
  __syncthreads();
  if (threadIdx.x < H_) {
    int h = threadIdx.x;
    float acc = br[h];
    for (int d = 0; d < D_; ++d) acc += row[d] * Wr[d * H_ + h];
    rel[((size_t)(b * H_ + h)) * S_ + s] = acc;
  }
}

// ---------------------------------------------------------------------------
// Fused attention: per (b, h, 16-query-row block):
//   scores = (1-g)*scale*QK^T + g*(scale*KGQ KGK^T + rel); mask -> NEG
//   softmax over full 2048-key row (row block lives in LDS)
//   write attn (f32, non-temporal b128) to d_out;
//   out = (1-g)*attn@V + g*attn@KGV -> out_heads
// 128 threads = 4 waves; waves split keys (phase 1) and head-dim (phase 2).
// ---------------------------------------------------------------------------
__global__ __launch_bounds__(128) void attn_kernel(
    const _Float16* __restrict__ Q,  const _Float16* __restrict__ Kk,
    const _Float16* __restrict__ V,  const _Float16* __restrict__ KQ,
    const _Float16* __restrict__ KK, const _Float16* __restrict__ KV,
    const float* __restrict__ rel, const float* __restrict__ gate,
    const int* __restrict__ mask,
    float* __restrict__ attn_out, float* __restrict__ out_heads) {
  extern __shared__ char smem[];
  float*    sc   = (float*)smem;            // 16 x 2048 scores/probs
  float*    relc = sc + 16 * S_;            // 2048 rel bias per key
  float*    mk   = relc + S_;               // 2048 mask per key
  float*    gq   = mk + S_;                 // 16 gate per query row
  float*    red  = gq + 16;                 // 16 x 8 reduction scratch
  float*    rowm = red + 128;               // 16 row max
  float*    rows = rowm + 16;               // 16 row sum
  _Float16* vsV  = (_Float16*)(rows + 16);  // 32 x 72 staged V tile
  _Float16* vsK  = vsV + 32 * 72;           // 32 x 72 staged KV tile

  const int tid  = threadIdx.x;
  const int lane = tid & 31;
  const int wave = tid >> 5;
  const int q0   = blockIdx.x * 16;
  const int h    = blockIdx.y;
  const int b    = blockIdx.z;

  // Stage per-key rel/mask and per-row gate
  for (int c = tid; c < S_; c += 128) {
    relc[c] = rel[((size_t)(b * H_ + h)) * S_ + c];
    mk[c]   = (float)mask[(size_t)b * S_ + c];
  }
  if (tid < 16) gq[tid] = gate[(size_t)b * S_ + q0 + tid];

  // Preload Q / KGQ fragments for this 16-row block (HD=64 -> 2 k-chunks)
  const _Float16* Qp  = Q  + ((size_t)(b * S_) + q0) * D_ + h * HD_;
  const _Float16* KQp = KQ + ((size_t)(b * S_) + q0) * D_ + h * HD_;
  v16h aQ0 = frag_a_f16(Qp,  D_, 0, 0, lane);
  v16h aQ1 = frag_a_f16(Qp,  D_, 0, 32, lane);
  v16h aG0 = frag_a_f16(KQp, D_, 0, 0, lane);
  v16h aG1 = frag_a_f16(KQp, D_, 0, 32, lane);
  __syncthreads();

  // ---- Phase 1: combined masked scores into LDS (waves split key tiles) ----
  const _Float16* Kp  = Kk + (size_t)b * S_ * D_ + h * HD_;
  const _Float16* KKp = KK + (size_t)b * S_ * D_ + h * HD_;
  for (int kt = wave; kt < S_ / 16; kt += 4) {
    v8f c = {}, ck = {};
    c  = wmma32(aQ0, frag_b_nmajor(Kp,  D_, 0,  kt * 16, lane), c);
    c  = wmma32(aQ1, frag_b_nmajor(Kp,  D_, 32, kt * 16, lane), c);
    ck = wmma32(aG0, frag_b_nmajor(KKp, D_, 0,  kt * 16, lane), ck);
    ck = wmma32(aG1, frag_b_nmajor(KKp, D_, 32, kt * 16, lane), ck);

    int col = kt * 16 + (lane & 15);
    float relv = relc[col];
    float mv   = mk[col];
#pragma unroll
    for (int r = 0; r < 8; ++r) {
      int row = r + ((lane >> 4) << 3);
      float g = gq[row];
      float v = (1.0f - g) * (c[r] * SCALE) + g * (ck[r] * SCALE + relv);
      if (mv == 0.0f) v = NEGV;
      sc[(size_t)row * S_ + col] = v;
    }
  }
  __syncthreads();

  // ---- Softmax over each 2048-wide row; NT-stream attn to global ----
  {
    int r   = tid >> 3;
    int seg = tid & 7;
    int c0  = seg * 256;
    float mx = -3.0e38f;
    for (int c = 0; c < 256; ++c) mx = fmaxf(mx, sc[(size_t)r * S_ + c0 + c]);
    red[r * 8 + seg] = mx;
    __syncthreads();
    if (seg == 0) {
      float m2 = red[r * 8];
      for (int i = 1; i < 8; ++i) m2 = fmaxf(m2, red[r * 8 + i]);
      rowm[r] = m2;
    }
    __syncthreads();
    float rm = rowm[r];
    float sum = 0.0f;
    for (int c = 0; c < 256; ++c) {
      float e = __expf(sc[(size_t)r * S_ + c0 + c] - rm);
      sc[(size_t)r * S_ + c0 + c] = e;
      sum += e;
    }
    red[r * 8 + seg] = sum;
    __syncthreads();
    if (seg == 0) {
      float s2 = 0.0f;
      for (int i = 0; i < 8; ++i) s2 += red[r * 8 + i];
      rows[r] = s2;
    }
    __syncthreads();
    float inv = 1.0f / rows[r];
    float* aout = attn_out + (((size_t)(b * H_ + h)) * S_ + q0 + r) * S_ + c0;
    // attn is write-once and never re-read by later kernels: use non-temporal
    // b128 stores so the 268MB stream doesn't evict K/V tiles from the 192MB L2.
    for (int c = 0; c < 256; c += 4) {
      f32x4 v4;
#pragma unroll
      for (int j = 0; j < 4; ++j) {
        float v = sc[(size_t)r * S_ + c0 + c + j] * inv;
        sc[(size_t)r * S_ + c0 + c + j] = v;
        v4[j] = v;
      }
      __builtin_nontemporal_store(v4, (f32x4*)(aout + c));
    }
  }
  __syncthreads();

  // ---- Phase 2: out = attn @ V, kg_out = attn @ KGV (waves split head dim) ----
  const _Float16* Vp  = V  + (size_t)b * S_ * D_ + h * HD_;
  const _Float16* KVp = KV + (size_t)b * S_ * D_ + h * HD_;
  const int n0w = wave * 16;
  v8f av = {}, ak = {};
  for (int kc = 0; kc < S_ / 32; ++kc) {
    // Stage 32 keys x 64 dims of V and KGV
    for (int i = tid; i < 32 * 64; i += 128) {
      int kk = i >> 6, d = i & 63;
      vsV[kk * 72 + d] = Vp[(size_t)(kc * 32 + kk) * D_ + d];
      vsK[kk * 72 + d] = KVp[(size_t)(kc * 32 + kk) * D_ + d];
    }
    __syncthreads();
    v16h aA = frag_a_f32(sc, S_, 0, kc * 32, lane);
    av = wmma32(aA, frag_b_kmajor(vsV, 72, 0, n0w, lane), av);
    ak = wmma32(aA, frag_b_kmajor(vsK, 72, 0, n0w, lane), ak);
    __syncthreads();
  }
#pragma unroll
  for (int r = 0; r < 8; ++r) {
    int row = r + ((lane >> 4) << 3);
    float g = gq[row];
    out_heads[((size_t)(b * S_) + q0 + row) * D_ + h * HD_ + n0w + (lane & 15)] =
        (1.0f - g) * av[r] + g * ak[r];
  }
}

// ---------------------------------------------------------------------------
// Host-side orchestration
// ---------------------------------------------------------------------------
extern "C" void kernel_launch(void* const* d_in, const int* in_sizes, int n_in,
                              void* d_out, int out_size, void* d_ws, size_t ws_size,
                              hipStream_t stream) {
  (void)in_sizes; (void)n_in; (void)out_size; (void)ws_size;
  const float* query = (const float*)d_in[0];
  const float* key   = (const float*)d_in[1];
  const float* value = (const float*)d_in[2];
  const float* kg    = (const float*)d_in[3];
  const float* relw  = (const float*)d_in[4];
  const int*   amask = (const int*)d_in[5];
  const float* Wq = (const float*)d_in[6],  *bq = (const float*)d_in[7];
  const float* Wk = (const float*)d_in[8],  *bk = (const float*)d_in[9];
  const float* Wv = (const float*)d_in[10], *bv = (const float*)d_in[11];
  const float* Wo = (const float*)d_in[12], *bo = (const float*)d_in[13];
  const float* Wkq = (const float*)d_in[14], *bkq = (const float*)d_in[15];
  const float* Wkk = (const float*)d_in[16], *bkk = (const float*)d_in[17];
  const float* Wkv = (const float*)d_in[18], *bkv = (const float*)d_in[19];
  const float* Wr = (const float*)d_in[20], *br = (const float*)d_in[21];
  const float* Wg = (const float*)d_in[22], *bg = (const float*)d_in[23];

  // Workspace carve-out
  char* ws = (char*)d_ws;
  size_t off = 0;
  auto carve = [&](size_t bytes) -> char* {
    char* p = ws + off;
    off += (bytes + 255) & ~(size_t)255;
    return p;
  };
  const size_t actH = (size_t)B_ * S_ * D_ * sizeof(_Float16);
  _Float16* Q16  = (_Float16*)carve(actH);
  _Float16* K16  = (_Float16*)carve(actH);
  _Float16* V16  = (_Float16*)carve(actH);
  _Float16* KQ16 = (_Float16*)carve(actH);
  _Float16* KK16 = (_Float16*)carve(actH);
  _Float16* KV16 = (_Float16*)carve(actH);
  float* gbuf  = (float*)carve((size_t)B_ * S_ * D_ * sizeof(float));
  float* gatev = (float*)carve((size_t)B_ * S_ * sizeof(float));
  float* relb  = (float*)carve((size_t)B_ * H_ * S_ * sizeof(float));
  float* outh  = (float*)carve((size_t)B_ * S_ * D_ * sizeof(float));

  float* outp  = (float*)d_out;                       // (B,S,D) fp32
  float* attnp = outp + (size_t)B_ * S_ * D_;         // (B,H,S,S) fp32

  dim3 blk(128);
  dim3 ggrid(D_ / 64, (B_ * S_) / 64);

  // Six projections -> f16
  gemm64<0><<<ggrid, blk, 0, stream>>>(query, nullptr, D_, D_, Wq,  D_, bq,  Q16,  nullptr, D_, D_);
  gemm64<0><<<ggrid, blk, 0, stream>>>(key,   nullptr, D_, D_, Wk,  D_, bk,  K16,  nullptr, D_, D_);
  gemm64<0><<<ggrid, blk, 0, stream>>>(value, nullptr, D_, D_, Wv,  D_, bv,  V16,  nullptr, D_, D_);
  gemm64<0><<<ggrid, blk, 0, stream>>>(query, nullptr, D_, D_, Wkq, D_, bkq, KQ16, nullptr, D_, D_);
  gemm64<0><<<ggrid, blk, 0, stream>>>(kg,    nullptr, D_, D_, Wkk, D_, bkk, KK16, nullptr, D_, D_);
  gemm64<0><<<ggrid, blk, 0, stream>>>(kg,    nullptr, D_, D_, Wkv, D_, bkv, KV16, nullptr, D_, D_);

  // Gate: sigmoid(concat(query,kg) @ Wg + bg), then row mean
  gemm64<2><<<ggrid, blk, 0, stream>>>(query, kg, D_, D_, Wg, D_, bg, nullptr, gbuf, D_, 2 * D_);
  gate_reduce<<<B_ * S_, 128, 0, stream>>>(gbuf, gatev);

  // Relation bias
  rel_kernel<<<B_ * S_, 64, 0, stream>>>(relw, Wr, br, relb);

  // Fused attention
  size_t shmem = (size_t)(16 * S_ + S_ + S_ + 16 + 128 + 16 + 16) * sizeof(float)
               + (size_t)(2 * 32 * 72) * sizeof(_Float16);
  dim3 agrid(S_ / 16, H_, B_);
  attn_kernel<<<agrid, blk, shmem, stream>>>(Q16, K16, V16, KQ16, KK16, KV16,
                                             relb, gatev, amask, attnp, outh);

  // Output projection -> d_out
  gemm64<1><<<ggrid, blk, 0, stream>>>(outh, nullptr, D_, D_, Wo, D_, bo, nullptr, outp, D_, D_);
}
